// SAGE_1709396984374
// MI455X (gfx1250) — matrix operand order repaired and checked by reading, not compiled, not run on hardware
//
#include <hip/hip_runtime.h>

// ---------------------------------------------------------------------------
// GraphSAGE 3-layer forward for MI455X (gfx1250, wave32, WMMA).
//  - scatter/gather: memory bound (~2.5GB total -> ~105us @ 23.3TB/s), f32
//    native global_atomic_add_f32 via inline asm.
//  - GEMM: [h | h_neigh] (bf16, K=256) x packed [W_self;W_neigh] via
//    v_wmma_f32_16x16x32_bf16, B staged in LDS (shared by 8 waves/block).
// ---------------------------------------------------------------------------

#define NNODES   50000
#define NEDGES   800000
#define FEATS    128
#define NCLASSES 47
#define MTILES   (NNODES / 16)   // 3125 exactly

typedef __attribute__((ext_vector_type(16))) __bf16 v16bf;
typedef __attribute__((ext_vector_type(8)))  float  v8f;

__device__ __forceinline__ void gatomic_add_f32(float* p, float v) {
  // Guarantee the native CDNA5 FP32 global atomic (no CAS loop).
  asm volatile("global_atomic_add_f32 %0, %1, off" :: "v"(p), "v"(v) : "memory");
}

__global__ void k_zero_f32(float* __restrict__ p, int n) {
  int i = blockIdx.x * blockDim.x + threadIdx.x;
  if (i < n) p[i] = 0.f;
}

__global__ void k_degree(const int* __restrict__ dst, float* __restrict__ deg, int n_edges) {
  int e = blockIdx.x * blockDim.x + threadIdx.x;
  if (e < n_edges) gatomic_add_f32(deg + dst[e], 1.0f);
}

// One wave per edge; each lane moves 4 contiguous floats (128 feats / 32 lanes).
__global__ void k_scatter(const float* __restrict__ h, const int* __restrict__ src,
                          const int* __restrict__ dst, float* __restrict__ agg, int n_edges) {
  int gid  = blockIdx.x * blockDim.x + threadIdx.x;
  int e    = gid >> 5;
  int lane = gid & 31;
  if (e >= n_edges) return;
  int s = src[e], d = dst[e];
  const float4 v = *(const float4*)(h + (size_t)s * FEATS + lane * 4);
  float* p = agg + (size_t)d * FEATS + lane * 4;
  gatomic_add_f32(p + 0, v.x);
  gatomic_add_f32(p + 1, v.y);
  gatomic_add_f32(p + 2, v.z);
  gatomic_add_f32(p + 3, v.w);
}

// A[node, 0:128] = bf16(h); A[node, 128:256] = bf16(agg / max(deg,1))
__global__ void k_build_a(const float* __restrict__ h, const float* __restrict__ agg,
                          const float* __restrict__ deg, __bf16* __restrict__ A, int n_nodes) {
  int gid = blockIdx.x * blockDim.x + threadIdx.x;
  if (gid >= n_nodes * FEATS) return;
  int node = gid >> 7, f = gid & 127;
  float dg = deg[node];
  dg = dg < 1.f ? 1.f : dg;
  A[(size_t)node * 256 + f]       = (__bf16)h[gid];
  A[(size_t)node * 256 + 128 + f] = (__bf16)(agg[gid] / dg);
}

// Pack [W_self; W_neigh] (K=256 x d_out) into WMMA B-fragment order:
// idx = ((kt*NT + nt)*32 + lane)*16 + j  holds  B[kt*32 + (lane&16?8:0) + (j&7) + (j&8?16:0)][nt*16 + (lane&15)]
__global__ void k_pack_w(const float* __restrict__ w_self, const float* __restrict__ w_neigh,
                         __bf16* __restrict__ Bp, int d_out, int NT) {
  int idx = blockIdx.x * blockDim.x + threadIdx.x;
  if (idx >= NT * 4096) return;
  int j    = idx & 15;
  int lane = (idx >> 4) & 31;
  int rest = idx >> 9;            // = kt*NT + nt
  int nt   = rest % NT;
  int kt   = rest / NT;
  int k = kt * 32 + ((lane & 16) ? 8 : 0) + (j & 7) + ((j & 8) ? 16 : 0);
  int n = nt * 16 + (lane & 15);
  float v = 0.f;
  if (n < d_out)
    v = (k < FEATS) ? w_self[k * d_out + n] : w_neigh[(k - FEATS) * d_out + n];
  Bp[idx] = (__bf16)v;
}

// out[M=16*mtile.., N=d_out] = A(bf16, K=256) @ Bp + bias.
// 8 waves/block, one M-tile per wave, B staged in LDS (shared by all waves).
template <int NT>
__global__ __launch_bounds__(256) void k_gemm(const __bf16* __restrict__ A,
                                              const __bf16* __restrict__ Bp,
                                              const float* __restrict__ bias,
                                              float* __restrict__ out,
                                              int n_tiles, int d_out) {
  extern __shared__ __align__(16) char smem_raw[];
  const int tid = threadIdx.x;

  // Stage packed B (NT*8192 bytes) into LDS as b128s.
  {
    uint4* s4 = (uint4*)smem_raw;
    const uint4* g4 = (const uint4*)Bp;
    for (int i = tid; i < NT * 512; i += 256) s4[i] = g4[i];
  }
  __syncthreads();

  const __bf16* sB = (const __bf16*)smem_raw;
  int wave = tid >> 5, lane = tid & 31;
  int mtile = blockIdx.x * 8 + wave;
  if (mtile >= n_tiles) return;   // wave-uniform exit, after the barrier

  // A-fragment base: lanes 0-15 hold K{0..7,16..23}, lanes 16-31 K{8..15,24..31}
  const __bf16* Arow = A + (size_t)(mtile * 16 + (lane & 15)) * 256 + ((lane & 16) ? 8 : 0);

  v8f acc[NT] = {};
#pragma unroll
  for (int kt = 0; kt < 8; ++kt) {
    union { uint4 u[2]; v16bf v; } af;
    af.u[0] = *(const uint4*)(Arow + kt * 32);
    af.u[1] = *(const uint4*)(Arow + kt * 32 + 16);
    if (kt < 7) __builtin_prefetch(Arow + (kt + 1) * 32, 0, 3);
#pragma unroll
    for (int nt = 0; nt < NT; ++nt) {
      union { uint4 u[2]; v16bf v; } bf_;
      const uint4* bp = (const uint4*)(sB + ((size_t)(kt * NT + nt) * 32 + lane) * 16);
      bf_.u[0] = bp[0];
      bf_.u[1] = bp[1];
      acc[nt] = __builtin_amdgcn_wmma_f32_16x16x32_bf16(
          false, af.v, false, bf_.v, (short)0, acc[nt], false, false);
    }
  }

  // C layout: VGPR j -> M = j (lanes 0-15) / 8+j (lanes 16-31), N = lane&15.
  int rbase = mtile * 16 + ((lane & 16) ? 8 : 0);
  int cl = lane & 15;
#pragma unroll
  for (int nt = 0; nt < NT; ++nt) {
    int col = nt * 16 + cl;
    if (col < d_out) {
      float bv = bias[col];
#pragma unroll
      for (int j = 0; j < 8; ++j)
        out[(size_t)(rbase + j) * d_out + col] = acc[nt][j] + bv;
    }
  }
}

// LayerNorm + ReLU, one wave per node (4 floats per lane), wave32 shuffles.
__global__ void k_ln_relu(const float* __restrict__ in, float* __restrict__ outp,
                          const float* __restrict__ g, const float* __restrict__ be,
                          int n_nodes) {
  int gid = blockIdx.x * blockDim.x + threadIdx.x;
  int node = gid >> 5, lane = gid & 31;
  if (node >= n_nodes) return;
  const float4 v = *(const float4*)(in + (size_t)node * FEATS + lane * 4);
  float s  = v.x + v.y + v.z + v.w;
  float ss = v.x * v.x + v.y * v.y + v.z * v.z + v.w * v.w;
  for (int off = 16; off > 0; off >>= 1) {
    s  += __shfl_xor(s,  off, 32);
    ss += __shfl_xor(ss, off, 32);
  }
  float mu  = s * (1.f / FEATS);
  float var = ss * (1.f / FEATS) - mu * mu;
  float rs  = rsqrtf(var + 1e-5f);
  int c = lane * 4;
  float4 o;
  o.x = fmaxf(0.f, (v.x - mu) * rs * g[c + 0] + be[c + 0]);
  o.y = fmaxf(0.f, (v.y - mu) * rs * g[c + 1] + be[c + 1]);
  o.z = fmaxf(0.f, (v.z - mu) * rs * g[c + 2] + be[c + 2]);
  o.w = fmaxf(0.f, (v.w - mu) * rs * g[c + 3] + be[c + 3]);
  *(float4*)(outp + (size_t)node * FEATS + lane * 4) = o;
}

extern "C" void kernel_launch(void* const* d_in, const int* in_sizes, int n_in,
                              void* d_out, int out_size, void* d_ws, size_t ws_size,
                              hipStream_t stream) {
  const float* x        = (const float*)d_in[0];
  const int*   src      = (const int*)d_in[1];
  const int*   dst      = (const int*)d_in[2];
  const float* w_self0  = (const float*)d_in[3];
  const float* w_neigh0 = (const float*)d_in[4];
  const float* b0       = (const float*)d_in[5];
  const float* g0       = (const float*)d_in[6];
  const float* be0      = (const float*)d_in[7];
  const float* w_self1  = (const float*)d_in[8];
  const float* w_neigh1 = (const float*)d_in[9];
  const float* b1       = (const float*)d_in[10];
  const float* g1       = (const float*)d_in[11];
  const float* be1      = (const float*)d_in[12];
  const float* w_self2  = (const float*)d_in[13];
  const float* w_neigh2 = (const float*)d_in[14];
  const float* b2       = (const float*)d_in[15];
  float* out = (float*)d_out;

  // Workspace layout (256B-aligned slices). tmp (GEMM out) aliases agg:
  // agg is dead once A is built, and it is re-zeroed next layer.
  char* ws = (char*)d_ws;
  size_t off = 0;
  auto take = [&](size_t bytes) {
    size_t o = off;
    off += (bytes + 255) & ~(size_t)255;
    return ws + o;
  };
  float*  deg  = (float*)take((size_t)NNODES * 4);
  float*  agg  = (float*)take((size_t)NNODES * FEATS * 4);   // also GEMM tmp out
  __bf16* Abuf = (__bf16*)take((size_t)NNODES * 256 * 2);
  float*  h0   = (float*)take((size_t)NNODES * FEATS * 4);
  float*  h1   = (float*)take((size_t)NNODES * FEATS * 4);
  __bf16* Bp0  = (__bf16*)take(8 * 4096 * 2);
  __bf16* Bp1  = (__bf16*)take(8 * 4096 * 2);
  __bf16* Bp2  = (__bf16*)take(3 * 4096 * 2);

  auto cdiv = [](int a, int b) { return (a + b - 1) / b; };
  const int feat_blocks = cdiv(NNODES * FEATS, 256);   // 25000
  const int scat_blocks = cdiv(NEDGES * 32, 256);      // 100000
  const int gemm_blocks = cdiv(MTILES, 8);             // 391

  // Degree (graph-invariant across layers) + weight packing.
  k_zero_f32<<<cdiv(NNODES, 256), 256, 0, stream>>>(deg, NNODES);
  k_degree<<<cdiv(NEDGES, 256), 256, 0, stream>>>(dst, deg, NEDGES);
  k_pack_w<<<cdiv(8 * 4096, 256), 256, 0, stream>>>(w_self0, w_neigh0, Bp0, FEATS, 8);
  k_pack_w<<<cdiv(8 * 4096, 256), 256, 0, stream>>>(w_self1, w_neigh1, Bp1, FEATS, 8);
  k_pack_w<<<cdiv(3 * 4096, 256), 256, 0, stream>>>(w_self2, w_neigh2, Bp2, NCLASSES, 3);

  // ---- Layer 0: x -> h0 ----
  k_zero_f32<<<feat_blocks, 256, 0, stream>>>(agg, NNODES * FEATS);
  k_scatter<<<scat_blocks, 256, 0, stream>>>(x, src, dst, agg, NEDGES);
  k_build_a<<<feat_blocks, 256, 0, stream>>>(x, agg, deg, Abuf, NNODES);
  k_gemm<8><<<gemm_blocks, 256, 8 * 8192, stream>>>(Abuf, Bp0, b0, agg, MTILES, FEATS);
  k_ln_relu<<<cdiv(NNODES * 32, 256), 256, 0, stream>>>(agg, h0, g0, be0, NNODES);

  // ---- Layer 1: h0 -> h1 ----
  k_zero_f32<<<feat_blocks, 256, 0, stream>>>(agg, NNODES * FEATS);
  k_scatter<<<scat_blocks, 256, 0, stream>>>(h0, src, dst, agg, NEDGES);
  k_build_a<<<feat_blocks, 256, 0, stream>>>(h0, agg, deg, Abuf, NNODES);
  k_gemm<8><<<gemm_blocks, 256, 8 * 8192, stream>>>(Abuf, Bp1, b1, agg, MTILES, FEATS);
  k_ln_relu<<<cdiv(NNODES * 32, 256), 256, 0, stream>>>(agg, h1, g1, be1, NNODES);

  // ---- Layer 2: h1 -> out [NNODES, 47] ----
  k_zero_f32<<<feat_blocks, 256, 0, stream>>>(agg, NNODES * FEATS);
  k_scatter<<<scat_blocks, 256, 0, stream>>>(h1, src, dst, agg, NEDGES);
  k_build_a<<<feat_blocks, 256, 0, stream>>>(h1, agg, deg, Abuf, NNODES);
  k_gemm<3><<<gemm_blocks, 256, 3 * 8192, stream>>>(Abuf, Bp2, b2, out, MTILES, NCLASSES);
}
